// GIN0_mn_10599979286633
// MI455X (gfx1250) — matrix operand order repaired and checked by reading, not compiled, not run on hardware
//
#include <hip/hip_runtime.h>
#include <hip/hip_bf16.h>

#define NN 50000
#define NE 800000
#define NG 500

typedef __attribute__((ext_vector_type(16))) __bf16 v16bf;
typedef __attribute__((ext_vector_type(8)))  float  v8f;

// ---------------- utility kernels ----------------

__global__ void zero_f32_kernel(float* __restrict__ p, int n) {
    int i = blockIdx.x * blockDim.x + threadIdx.x;
    if (i < n) p[i] = 0.0f;
}

// W: [K][M] row-major f32  ->  Wt: [M][K] bf16 (transposed)
__global__ void cvt_wt_kernel(const float* __restrict__ W, __bf16* __restrict__ Wt,
                              int K, int M) {
    int i = blockIdx.x * blockDim.x + threadIdx.x;
    if (i < K * M) {
        int m = i / K, k = i % K;
        Wt[i] = (__bf16)W[k * M + m];
    }
}

// out_bf16 = bf16(h + agg)
__global__ void hsum_bf16_kernel(const float* __restrict__ h, const float* __restrict__ agg,
                                 __bf16* __restrict__ out, int n) {
    int i = blockIdx.x * blockDim.x + threadIdx.x;
    if (i < n) out[i] = (__bf16)(h[i] + agg[i]);
}

// ---------------- fused edge-proj + gather + relu + scatter-add ----------------
// proj = edge_attr[e] @ We + be (16 x D matvec, We in LDS)
// m = relu(h[src] + proj); agg[dst] += m   (hw f32 atomics, L2-resident)
__global__ void message_kernel(const float* __restrict__ h,
                               const float* __restrict__ ea,
                               const int*   __restrict__ src,
                               const int*   __restrict__ dst,
                               const float* __restrict__ We,
                               const float* __restrict__ be,
                               float* __restrict__ agg, int D) {
    extern __shared__ float lds[];
    float* Wl = lds;            // 16*D
    float* bl = lds + 16 * D;   // D
    for (int i = threadIdx.x; i < 16 * D; i += blockDim.x) Wl[i] = We[i];
    for (int i = threadIdx.x; i < D;      i += blockDim.x) bl[i] = be[i];
    __syncthreads();

    int tpe = D >> 2;                       // threads per edge (float4 per thread)
    int epb = blockDim.x / tpe;             // edges per block
    int le  = threadIdx.x / tpe;
    int t   = threadIdx.x - le * tpe;
    int e   = blockIdx.x * epb + le;
    if (e >= NE) return;

    int c0 = t * 4;
    float a0 = bl[c0], a1 = bl[c0 + 1], a2 = bl[c0 + 2], a3 = bl[c0 + 3];
    const float* eap = ea + (size_t)e * 16;
#pragma unroll
    for (int k = 0; k < 16; ++k) {
        float av = eap[k];
        const float* w = Wl + k * D + c0;
        a0 = fmaf(av, w[0], a0); a1 = fmaf(av, w[1], a1);
        a2 = fmaf(av, w[2], a2); a3 = fmaf(av, w[3], a3);
    }
    const float* xs = h + (size_t)src[e] * D + c0;
    a0 = fmaxf(a0 + xs[0], 0.0f); a1 = fmaxf(a1 + xs[1], 0.0f);
    a2 = fmaxf(a2 + xs[2], 0.0f); a3 = fmaxf(a3 + xs[3], 0.0f);
    float* ap = agg + (size_t)dst[e] * D + c0;
    unsafeAtomicAdd(ap + 0, a0); unsafeAtomicAdd(ap + 1, a1);
    unsafeAtomicAdd(ap + 2, a2); unsafeAtomicAdd(ap + 3, a3);
}

// ---------------- WMMA GEMM: out = act(A[N][K] @ Wt[M][K]^T + bias) ----------------
// one 16x16 tile per wave; relu always applied (both MLP stages feed a relu)
template <bool OUT_BF16>
__global__ void gemm_wmma_kernel(const __bf16* __restrict__ A,
                                 const __bf16* __restrict__ Wt,
                                 const float*  __restrict__ bias,
                                 void* __restrict__ outv, int K, int M) {
    int wave = threadIdx.x >> 5;
    int lane = threadIdx.x & 31;
    int half = lane >> 4;          // 0: lanes 0-15, 1: lanes 16-31
    int l15  = lane & 15;
    int r0 = blockIdx.x * 16;
    int c0 = (blockIdx.y * (blockDim.x >> 5) + wave) * 16;

    // A fragment base: row = r0+l15; lane-half selects K subset {0..7,16..23} vs {8..15,24..31}
    const unsigned* aptr = (const unsigned*)(A + (size_t)(r0 + l15) * K + (half ? 8 : 0));
    // B fragment base: col = c0+l15; lane-half selects K subset {0..7,16..23} vs {8..15,24..31}
    const unsigned* bptr = (const unsigned*)(Wt + (size_t)(c0 + l15) * K + (half ? 8 : 0));

    v8f acc = {};
    union { unsigned u[8]; v16bf v; } af, bfu;
    for (int k0 = 0; k0 < K; k0 += 32) {
        const unsigned* ap = aptr + (k0 >> 1);
#pragma unroll
        for (int i = 0; i < 4; ++i) { af.u[i] = ap[i]; af.u[4 + i] = ap[8 + i]; }
        const unsigned* bp = bptr + (k0 >> 1);
#pragma unroll
        for (int i = 0; i < 4; ++i) { bfu.u[i] = bp[i]; bfu.u[4 + i] = bp[8 + i]; }
        acc = __builtin_amdgcn_wmma_f32_16x16x32_bf16(
            false, af.v, false, bfu.v, (short)0, acc, false, false);
    }

    int row = r0 + (half ? 8 : 0);
    int col = c0 + l15;
    float b = bias[col];
#pragma unroll
    for (int j = 0; j < 8; ++j) {
        float v = fmaxf(acc[j] + b, 0.0f);
        size_t off = (size_t)(row + j) * M + col;
        if (OUT_BF16) ((__bf16*)outv)[off] = (__bf16)v;
        else          ((float*)outv)[off]  = v;
    }
}

// ---------------- batch norm ----------------
__global__ void bn_stats_kernel(const float* __restrict__ z, float* __restrict__ stats, int D) {
    int c  = threadIdx.x;              // blockDim.x == D
    int r0 = blockIdx.x * 128;
    int r1 = min(r0 + 128, NN);
    float s = 0.0f, sq = 0.0f;
    for (int r = r0; r < r1; ++r) {
        float v = z[(size_t)r * D + c];
        s += v; sq += v * v;
    }
    unsafeAtomicAdd(&stats[c], s);
    unsafeAtomicAdd(&stats[D + c], sq);
}

__global__ void bn_norm_kernel(const float* __restrict__ z, const float* __restrict__ stats,
                               const float* __restrict__ g, const float* __restrict__ bt,
                               float* __restrict__ out, int D) {
    size_t i = (size_t)blockIdx.x * blockDim.x + threadIdx.x;
    if (i >= (size_t)NN * D) return;
    int c = (int)(i % D);
    const float inv_n = 1.0f / (float)NN;
    float mu  = stats[c] * inv_n;
    float var = stats[D + c] * inv_n - mu * mu;
    out[i] = (z[i] - mu) * rsqrtf(var + 1e-5f) * g[c] + bt[c];
}

// ---------------- head ----------------
__global__ void head_kernel(const float* __restrict__ h, const int* __restrict__ nn,
                            const float* __restrict__ Wf1, const float* __restrict__ bf1,
                            const float* __restrict__ Wf2, const float* __restrict__ bf2,
                            float* __restrict__ out) {
    int g = blockIdx.x * blockDim.x + threadIdx.x;
    if (g >= NG) return;
    int last = -1;
    for (int i = 0; i <= g; ++i) last += nn[i];
    const float* m = h + (size_t)last * 64;
    float acc = bf2[0];
    for (int j = 0; j < 16; ++j) {
        float hj = bf1[j];
#pragma unroll
        for (int c = 0; c < 64; ++c) hj = fmaf(m[c], Wf1[c * 16 + j], hj);
        acc = fmaf(fmaxf(hj, 0.0f), Wf2[j], acc);
    }
    out[g] = acc;
}

// ---------------- host orchestration ----------------
extern "C" void kernel_launch(void* const* d_in, const int* in_sizes, int n_in,
                              void* d_out, int out_size, void* d_ws, size_t ws_size,
                              hipStream_t stream) {
    const float* x    = (const float*)d_in[0];
    const float* ea   = (const float*)d_in[1];
    const int*   eidx = (const int*)d_in[2];
    const int*   nnod = (const int*)d_in[3];
    const int* src = eidx;            // edge_index[0]
    const int* dst = eidx + NE;       // edge_index[1]

    struct Layer { const float *We, *be, *W1, *b1, *W2, *b2, *g, *bt; int Din, Dmid, Dout; };
    Layer L[3] = {
        { (const float*)d_in[4],  (const float*)d_in[5],  (const float*)d_in[6],  (const float*)d_in[7],
          (const float*)d_in[8],  (const float*)d_in[9],  (const float*)d_in[10], (const float*)d_in[11],
          128, 256, 256 },
        { (const float*)d_in[12], (const float*)d_in[13], (const float*)d_in[14], (const float*)d_in[15],
          (const float*)d_in[16], (const float*)d_in[17], (const float*)d_in[18], (const float*)d_in[19],
          256, 128, 128 },
        { (const float*)d_in[20], (const float*)d_in[21], (const float*)d_in[22], (const float*)d_in[23],
          (const float*)d_in[24], (const float*)d_in[25], (const float*)d_in[26], (const float*)d_in[27],
          128, 64, 64 },
    };
    const float* Wf1 = (const float*)d_in[28];
    const float* bf1 = (const float*)d_in[29];
    const float* Wf2 = (const float*)d_in[30];
    const float* bf2 = (const float*)d_in[31];

    // workspace carve-up (256B aligned)
    char* ws = (char*)d_ws;
    auto take = [&](size_t bytes) { char* p = ws; ws += (bytes + 255) & ~(size_t)255; return p; };
    float*  agg   = (float*)take((size_t)NN * 256 * 4);
    float*  hcur  = (float*)take((size_t)NN * 256 * 4);
    float*  zbuf  = (float*)take((size_t)NN * 256 * 4);
    __bf16* abf   = (__bf16*)take((size_t)NN * 256 * 2);
    __bf16* mbf   = (__bf16*)take((size_t)NN * 256 * 2);
    __bf16* wt1   = (__bf16*)take((size_t)256 * 256 * 2);
    __bf16* wt2   = (__bf16*)take((size_t)256 * 256 * 2);
    float*  stats = (float*)take(512 * 4);

    auto launch_gemm = [&](const __bf16* A, const __bf16* Wt, const float* bias,
                           void* out, int K, int M, bool bf16out) {
        int wpb = (M >= 128) ? 8 : (M / 16);
        dim3 blk(32 * wpb), grd(NN / 16, M / (16 * wpb));
        if (bf16out)
            gemm_wmma_kernel<true ><<<grd, blk, 0, stream>>>(A, Wt, bias, out, K, M);
        else
            gemm_wmma_kernel<false><<<grd, blk, 0, stream>>>(A, Wt, bias, out, K, M);
    };

    const float* hin = x;
    for (int l = 0; l < 3; ++l) {
        const Layer& P = L[l];
        int D = P.Din;
        // 1) agg = 0
        int na = NN * D;
        zero_f32_kernel<<<(na + 255) / 256, 256, 0, stream>>>(agg, na);
        // 2) fused edge-proj + message + scatter-add
        int epb = 1024 / D;
        size_t ldsb = (size_t)(16 * D + D) * 4;
        message_kernel<<<(NE + epb - 1) / epb, 256, ldsb, stream>>>(
            hin, ea, src, dst, P.We, P.be, agg, D);
        // 3) bf16 stage of (h + agg)
        hsum_bf16_kernel<<<(na + 255) / 256, 256, 0, stream>>>(hin, agg, abf, na);
        // 4) weights -> transposed bf16
        int n1 = P.Din * P.Dmid, n2 = P.Dmid * P.Dout;
        cvt_wt_kernel<<<(n1 + 255) / 256, 256, 0, stream>>>(P.W1, wt1, P.Din, P.Dmid);
        cvt_wt_kernel<<<(n2 + 255) / 256, 256, 0, stream>>>(P.W2, wt2, P.Dmid, P.Dout);
        // 5) MLP stage 1: relu(hsum @ W1 + b1) -> bf16
        launch_gemm(abf, wt1, P.b1, mbf, P.Din, P.Dmid, true);
        // 6) MLP stage 2 + relu (relu precedes BN in reference) -> f32
        launch_gemm(mbf, wt2, P.b2, zbuf, P.Dmid, P.Dout, false);
        // 7) batch norm
        zero_f32_kernel<<<1, 2 * P.Dout, 0, stream>>>(stats, 2 * P.Dout);
        bn_stats_kernel<<<(NN + 127) / 128, P.Dout, 0, stream>>>(zbuf, stats, P.Dout);
        int nb = NN * P.Dout;
        bn_norm_kernel<<<(nb + 255) / 256, 256, 0, stream>>>(
            zbuf, stats, P.g, P.bt, hcur, P.Dout);
        hin = hcur;
    }

    head_kernel<<<2, 256, 0, stream>>>(hcur, nnod, Wf1, bf1, Wf2, bf2, (float*)d_out);
}
